// DeepSNN_62225486185104
// MI455X (gfx1250) — matrix operand order repaired and checked by reading, not compiled, not run on hardware
//
#include <hip/hip_runtime.h>

// CDNA5 / gfx1250 fused SNN forward:
//   cur1 = x@W1^T + b1  (split-bf16 WMMA, ~fp32 accuracy, x streamed exactly once)
//   T x { mem1 = lif(mem1,cur1); cur2 = mem1@W2^T+b2 (WMMA); mem2 = lif(mem2,cur2) }
//   out = mem2@W3^T + b3
// One wave owns 16 batch rows; mem1/mem2/cur1 register resident for all steps.
// W1 chunk is converted to bf16 hi/lo cooperatively (256 threads) into LDS in
// fragment-ready layout (B-frags = 2x ds_load_b128). Tail K-chunk is peeled so
// the hot loop has zero bounds checks / EXEC-mask games.

typedef __attribute__((ext_vector_type(16))) __bf16 v16bf;
typedef __attribute__((ext_vector_type(8)))  __bf16 v8bf;
typedef __attribute__((ext_vector_type(8)))  float  v8f;

#define BETA   0.9f
#define THRESH 1.0f
#define KDIM   1700
#define NCHUNK ((KDIM + 31) / 32)   // 54 (last chunk zero-padded)

__device__ __forceinline__ void wait_lds() {
  // intra-wave DS store->load ordering (per-wave private LDS slab)
  asm volatile("s_wait_dscnt 0x0" ::: "memory");
}

__device__ __forceinline__ float4 ld4_guard(const float* p, int k, int K) {
  if (k + 4 <= K) return *(const float4*)p;
  float4 v; v.x = v.y = v.z = v.w = 0.f;
  if (k     < K) v.x = p[0];
  if (k + 1 < K) v.y = p[1];
  if (k + 2 < K) v.z = p[2];
  if (k + 3 < K) v.w = p[3];
  return v;
}

__device__ __forceinline__ v8f wmma_bf16(v16bf a, v16bf b, v8f c) {
  return __builtin_amdgcn_wmma_f32_16x16x32_bf16(false, a, false, b,
                                                 (short)0, c, false, false);
}

__device__ __forceinline__ v16bf cat8(v8bf a, v8bf b) {
  return __builtin_shufflevector(a, b, 0, 1, 2, 3, 4, 5, 6, 7,
                                       8, 9, 10, 11, 12, 13, 14, 15);
}

__global__ __launch_bounds__(256, 1) void snn_fused(
    const float* __restrict__ x,  const float* __restrict__ W1,
    const float* __restrict__ b1, const float* __restrict__ W2,
    const float* __restrict__ b2, const float* __restrict__ W3,
    const float* __restrict__ b3, const int* __restrict__ nsteps,
    float* __restrict__ out) {
  // per-wave private f32 slab (max use: 16x68 transpose region)
  __shared__ float slab_all[8][1104];
  // shared fragment-ready bf16 W1 chunk (64 neurons x 32 K), hi and lo parts
  __shared__ __align__(16) __bf16 w1h[64 * 32];
  __shared__ __align__(16) __bf16 w1l[64 * 32];

  const int wave = threadIdx.x >> 5;
  const int lane = threadIdx.x & 31;
  const int hi   = lane >> 4;        // lane-half select
  const int l16  = lane & 15;
  const int row0 = blockIdx.x * 128 + wave * 16;   // 16 batch rows per wave
  float* slab = slab_all[wave];

  // A-frag K index for element e (16-bit A 16x32 layout, CDNA5 ISA)
  auto kmap = [&](int e) { return hi * 8 + e + ((e >= 8) ? 8 : 0); };

  // ---------------- Phase 1: cur1 = x @ W1^T (+b1), split-bf16 WMMA ----------
  v8f acc[4];
  #pragma unroll
  for (int t = 0; t < 4; ++t) acc[t] = (v8f){0, 0, 0, 0, 0, 0, 0, 0};

  auto chunk_body = [&](int k0, bool tail) __attribute__((always_inline)) {
    // stage x[row0..row0+15][k0..k0+31] coalesced into per-wave slab (stride 36)
    {
      const int r     = lane >> 1;
      const int cbase = (lane & 1) * 16;
      const float* xr = x + (size_t)(row0 + r) * KDIM + k0 + cbase;
      #pragma unroll
      for (int j = 0; j < 4; ++j) {
        float4 v;
        if (!tail) v = *(const float4*)(xr + j * 4);
        else       v = ld4_guard(xr + j * 4, k0 + cbase + j * 4, KDIM);
        *(float4*)&slab[r * 36 + cbase + j * 4] = v;
      }
      if (!tail) __builtin_prefetch(xr + 32, 0, 1);   // global_prefetch_b8
    }
    // cooperative W1 chunk convert: 64x32 f32 -> bf16 hi/lo, 8 elems per thread
    {
      const int idx = threadIdx.x * 8;       // 256*8 = 2048 = 64*32
      const int n   = idx >> 5;
      const int kk  = idx & 31;
      const float* wr = W1 + (size_t)n * KDIM + k0 + kk;
      float4 a, b;
      if (!tail) {
        a = *(const float4*)wr;
        b = *(const float4*)(wr + 4);
      } else {
        a = ld4_guard(wr,     k0 + kk,     KDIM);
        b = ld4_guard(wr + 4, k0 + kk + 4, KDIM);
      }
      float wv[8] = {a.x, a.y, a.z, a.w, b.x, b.y, b.z, b.w};
      #pragma unroll
      for (int j = 0; j < 8; ++j) {
        __bf16 h = (__bf16)wv[j];
        w1h[idx + j] = h;
        w1l[idx + j] = (__bf16)(wv[j] - (float)h);
      }
    }
    __syncthreads();
    // A fragments (hi/lo split) from LDS; adjacent-K float2 reads
    v16bf ah, al;
    #pragma unroll
    for (int e = 0; e < 16; e += 2) {
      float2 f2 = *(const float2*)&slab[l16 * 36 + kmap(e)];
      __bf16 h0 = (__bf16)f2.x, h1 = (__bf16)f2.y;
      ah[e] = h0;                 ah[e + 1] = h1;
      al[e] = (__bf16)(f2.x - (float)h0);
      al[e + 1] = (__bf16)(f2.y - (float)h1);
    }
    // B fragments: batch all 4 tiles (h+l) first so WMMAs overlap the ds loads
    v16bf bh[4], bl[4];
    #pragma unroll
    for (int t = 0; t < 4; ++t) {
      const int off = (t * 16 + l16) * 32 + hi * 8;
      bh[t] = cat8(*(const v8bf*)&w1h[off], *(const v8bf*)&w1h[off + 16]);
      bl[t] = cat8(*(const v8bf*)&w1l[off], *(const v8bf*)&w1l[off + 16]);
    }
    #pragma unroll
    for (int t = 0; t < 4; ++t) {
      acc[t] = wmma_bf16(ah, bh[t], acc[t]);   // Ah*Bh
      acc[t] = wmma_bf16(al, bh[t], acc[t]);   // Al*Bh
      acc[t] = wmma_bf16(ah, bl[t], acc[t]);   // Ah*Bl
    }
    __syncthreads();   // protect w1h/w1l before next chunk overwrites
  };

  #pragma unroll 1
  for (int chunk = 0; chunk < NCHUNK - 1; ++chunk)   // chunks 0..52: no guards
    chunk_body(chunk * 32, false);
  chunk_body((NCHUNK - 1) * 32, true);               // peeled zero-padded tail

  // + b1 (D-layout: lane's column N = t*16+l16, bias depends on N only)
  #pragma unroll
  for (int t = 0; t < 4; ++t) {
    const float bv = b1[t * 16 + l16];
    #pragma unroll
    for (int i = 0; i < 8; ++i) acc[t][i] += bv;
  }

  // -------- transpose cur1 D-layout -> A-layout once (via LDS, stride 68) ----
  #pragma unroll
  for (int t = 0; t < 4; ++t)
    #pragma unroll
    for (int i = 0; i < 8; ++i)
      slab[(i + hi * 8) * 68 + t * 16 + l16] = acc[t][i];
  wait_lds();
  float cur1[32], mem1[32];
  #pragma unroll
  for (int c = 0; c < 2; ++c)
    #pragma unroll
    for (int e = 0; e < 16; e += 2) {
      float2 f2 = *(const float2*)&slab[l16 * 68 + c * 32 + kmap(e)];
      cur1[c * 16 + e]     = f2.x;
      cur1[c * 16 + e + 1] = f2.y;
      mem1[c * 16 + e]     = 0.f;
      mem1[c * 16 + e + 1] = 0.f;
    }

  // -------- W2 fragments (register resident, hi/lo split), b2 per lane -------
  v16bf w2h[2][2], w2l[2][2];
  float b2v[2];
  #pragma unroll
  for (int t2 = 0; t2 < 2; ++t2) {
    const int j = t2 * 16 + l16;
    b2v[t2] = b2[j];
    #pragma unroll
    for (int c = 0; c < 2; ++c) {
      const float* wr = W2 + j * 64 + c * 32 + hi * 8;
      #pragma unroll
      for (int e = 0; e < 16; ++e) {
        float  f = wr[e + ((e >= 8) ? 8 : 0)];
        __bf16 h = (__bf16)f;
        w2h[t2][c][e] = h;
        w2l[t2][c][e] = (__bf16)(f - (float)h);
      }
    }
  }

  // ---------------- Phase 2: T-step LIF recurrence, all in registers ---------
  v8f m2[2];
  m2[0] = (v8f){0, 0, 0, 0, 0, 0, 0, 0};
  m2[1] = (v8f){0, 0, 0, 0, 0, 0, 0, 0};
  const int T = nsteps[0];
  #pragma unroll 1
  for (int s = 0; s < T; ++s) {
    // mem1 = BETA*mem1 + cur1 - (mem1>thr)*thr   (A-layout resident)
    #pragma unroll
    for (int i = 0; i < 32; ++i) {
      const float m = mem1[i];
      mem1[i] = BETA * m + cur1[i] - (m > THRESH ? THRESH : 0.f);
    }
    // A frags from registers (no LDS in the loop)
    v16bf ah[2], al[2];
    #pragma unroll
    for (int c = 0; c < 2; ++c)
      #pragma unroll
      for (int e = 0; e < 16; ++e) {
        float  f = mem1[c * 16 + e];
        __bf16 h = (__bf16)f;
        ah[c][e] = h;
        al[c][e] = (__bf16)(f - (float)h);
      }
    // cur2 = mem1 @ W2^T + b2 ; mem2 LIF update
    #pragma unroll
    for (int t2 = 0; t2 < 2; ++t2) {
      v8f cc;
      #pragma unroll
      for (int i = 0; i < 8; ++i) cc[i] = b2v[t2];
      #pragma unroll
      for (int c = 0; c < 2; ++c) {
        cc = wmma_bf16(ah[c], w2h[t2][c], cc);
        cc = wmma_bf16(al[c], w2h[t2][c], cc);
        cc = wmma_bf16(ah[c], w2l[t2][c], cc);
      }
      #pragma unroll
      for (int i = 0; i < 8; ++i) {
        const float m = m2[t2][i];
        m2[t2][i] = BETA * m + cc[i] - (m > THRESH ? THRESH : 0.f);
      }
    }
  }

  // ---------------- Final: out = mem2 @ W3^T + b3 (tiny, via LDS) ------------
  #pragma unroll
  for (int t2 = 0; t2 < 2; ++t2)
    #pragma unroll
    for (int i = 0; i < 8; ++i)
      slab[(i + hi * 8) * 36 + t2 * 16 + l16] = m2[t2][i];
  wait_lds();
  {
    const int r = l16, o = hi;      // 16 rows x 2 outputs = 32 lanes
    float s = b3[o];
    #pragma unroll
    for (int n = 0; n < 32; ++n) s += slab[r * 36 + n] * W3[o * 32 + n];
    out[(size_t)(row0 + r) * 2 + o] = s;
  }
}

extern "C" void kernel_launch(void* const* d_in, const int* in_sizes, int n_in,
                              void* d_out, int out_size, void* d_ws, size_t ws_size,
                              hipStream_t stream) {
  const float* x      = (const float*)d_in[0];
  const float* W1     = (const float*)d_in[1];
  const float* b1     = (const float*)d_in[2];
  const float* W2     = (const float*)d_in[3];
  const float* b2     = (const float*)d_in[4];
  const float* W3     = (const float*)d_in[5];
  const float* b3     = (const float*)d_in[6];
  const int*   nsteps = (const int*)d_in[7];
  const int B = in_sizes[0] / KDIM;      // 16384
  const int blocks = B / 128;            // 128 rows per block (8 waves x 16 rows)
  snn_fused<<<blocks, 256, 0, stream>>>(x, W1, b1, W2, b2, W3, b3, nsteps,
                                        (float*)d_out);
}
